// ClassicalQLSTM_65481071410792
// MI455X (gfx1250) — matrix operand order repaired and checked by reading, not compiled, not run on hardware
//
#include <hip/hip_runtime.h>
#include <hip/hip_bf16.h>

typedef __bf16 bf16_t;
typedef __attribute__((ext_vector_type(16))) __bf16 v16bf;
typedef __attribute__((ext_vector_type(8)))  __bf16 v8bf;
typedef __attribute__((ext_vector_type(8)))  float   v8f;
typedef __attribute__((ext_vector_type(4)))  float   v4f;
typedef __attribute__((ext_vector_type(4)))  int     v4i;

#define T_STEPS 1024
#define BATCH   64
#define DH      256
#define NCOL    1024          // 4 gates * DH
#define CHUNK_T 128
#define NCHUNK  (T_STEPS / CHUNK_T)
#define CHUNK_ROWS (CHUNK_T * BATCH)   // 8192

// ---------------------------------------------------------------------------
// Weight pre-pack: f32 [512,256] row-major -> bf16 [col=g*256+n][k] (transposed)
// so B-fragments (32x16, K-major per column) are contiguous 16B loads.
// ---------------------------------------------------------------------------
__global__ void pack_weights_kernel(const float* __restrict__ Wf, const float* __restrict__ Wi,
                                    const float* __restrict__ Wg, const float* __restrict__ Wo,
                                    const float* __restrict__ bf_, const float* __restrict__ bi_,
                                    const float* __restrict__ bg_, const float* __restrict__ bo_,
                                    bf16_t* __restrict__ Wxp, bf16_t* __restrict__ Whp,
                                    float* __restrict__ bias) {
    int idx = blockIdx.x * blockDim.x + threadIdx.x;   // 0 .. NCOL*DH-1
    if (idx >= NCOL * DH) return;
    int col = idx >> 8;          // 0..1023  (g*256 + n)
    int k   = idx & 255;         // 0..255
    int g   = col >> 8;
    int n   = col & 255;
    const float* W = (g == 0) ? Wf : (g == 1) ? Wi : (g == 2) ? Wg : Wo;
    const float* b = (g == 0) ? bf_ : (g == 1) ? bi_ : (g == 2) ? bg_ : bo_;
    Wxp[idx] = (bf16_t)W[k * DH + n];            // x-part rows 0..255
    Whp[idx] = (bf16_t)W[(DH + k) * DH + n];     // h-part rows 256..511
    if (k == 0) bias[col] = b[n];
}

__global__ void zero_state_kernel(bf16_t* __restrict__ hstate, float* __restrict__ cstate) {
    int i = blockIdx.x * blockDim.x + threadIdx.x;
    if (i < BATCH * DH) { hstate[i] = (bf16_t)0.0f; cstate[i] = 0.0f; }
}

// ---------------------------------------------------------------------------
// Per-chunk X f32 -> bf16 (one pass; removes per-tile cvt + redundant loads)
// ---------------------------------------------------------------------------
__global__ __launch_bounds__(256) void xconv_kernel(const float* __restrict__ X,
                                                    bf16_t* __restrict__ Xb) {
    int i = blockIdx.x * blockDim.x + threadIdx.x;     // one thread per 8 elements
    const float* src = X + i * 8;
    v4f a0, a1;
#pragma unroll
    for (int j = 0; j < 4; ++j) a0[j] = __builtin_nontemporal_load(src + j);
#pragma unroll
    for (int j = 0; j < 4; ++j) a1[j] = __builtin_nontemporal_load(src + 4 + j);
    v8bf o;
#pragma unroll
    for (int j = 0; j < 4; ++j) { o[j] = (bf16_t)a0[j]; o[4 + j] = (bf16_t)a1[j]; }
    *(v8bf*)(Xb + i * 8) = o;
}

// ---------------------------------------------------------------------------
// Phase 1: Z[r][col] = Xb[r] . Wx[:,col] + bias[col]   (bf16 WMMA, f32 accum)
// A panel (16 rows x 256 k, 8KB) staged once per block in LDS via async-to-LDS.
// ---------------------------------------------------------------------------
__global__ __launch_bounds__(256) void xproj_kernel(const bf16_t* __restrict__ Xb,     // [rows][256] bf16
                                                    const bf16_t* __restrict__ Wxp,    // [NCOL][DH]
                                                    const float* __restrict__ bias,    // [NCOL]
                                                    float* __restrict__ Z) {           // [rows][NCOL]
    __shared__ bf16_t apanel[16 * DH];                 // 8 KB, contiguous rows
    const int tid   = threadIdx.x;
    const int lane  = tid & 31;
    const int wave  = tid >> 5;
    const int mtile = blockIdx.x;
    const int ntile = blockIdx.y * 8 + wave;           // 0..63
    const int l16   = lane & 15;
    const int hi    = lane >> 4;                       // 0 | 1
    const int col   = ntile * 16 + l16;

    // ---- stage A panel (8192 contiguous bytes) -----------------------------
    const bf16_t* gsrc = Xb + (size_t)mtile * 16 * DH;
#if __has_builtin(__builtin_amdgcn_global_load_async_to_lds_b128) && \
    __has_builtin(__builtin_amdgcn_s_wait_asynccnt)
    {
        typedef __attribute__((address_space(1))) v4i gv4i;
        typedef __attribute__((address_space(3))) v4i lv4i;
        gv4i* g0 = (gv4i*)gsrc;      // addrspacecast generic->global, reinterpret to int4
        lv4i* l0 = (lv4i*)apanel;    // addrspacecast generic->lds,    reinterpret to int4
        __builtin_amdgcn_global_load_async_to_lds_b128(g0 + tid,       l0 + tid,       0, 0);
        __builtin_amdgcn_global_load_async_to_lds_b128(g0 + tid + 256, l0 + tid + 256, 0, 0);
        __builtin_amdgcn_s_wait_asynccnt(0);
    }
#else
    {
        v8bf*       dst = (v8bf*)apanel;
        const v8bf* src = (const v8bf*)gsrc;
        dst[tid]       = src[tid];
        dst[tid + 256] = src[tid + 256];
    }
#endif
    __syncthreads();

    const bf16_t* aBase = apanel + l16 * DH + hi * 8;
    const bf16_t* bBase = Wxp + col * DH + hi * 8;

    v8f acc = {};
#pragma unroll
    for (int kk = 0; kk < 8; ++kk) {
        const int k0 = kk * 32;
        // A fragment: 16x32 bf16; lane<16: K {0..7,16..23}, lane>=16: K {8..15,24..31}
        v8bf aL0 = *(const v8bf*)(aBase + k0);
        v8bf aL1 = *(const v8bf*)(aBase + k0 + 16);
        v16bf a = __builtin_shufflevector(aL0, aL1, 0,1,2,3,4,5,6,7,8,9,10,11,12,13,14,15);
        // B fragment: 32x16 bf16; lane<16: K {0..7,16..23}, lane>=16: K {8..15,24..31}
        v8bf b0 = *(const v8bf*)(bBase + k0);
        v8bf b1 = *(const v8bf*)(bBase + k0 + 16);
        v16bf b = __builtin_shufflevector(b0, b1, 0,1,2,3,4,5,6,7,8,9,10,11,12,13,14,15);
        acc = __builtin_amdgcn_wmma_f32_16x16x32_bf16(false, a, false, b, (short)0, acc, false, false);
    }
    const float bcol    = bias[col];
    const int   rowBase = mtile * 16 + hi * 8;         // D layout: VGPR j -> row rowBase+j
#pragma unroll
    for (int j = 0; j < 8; ++j)
        Z[(rowBase + j) * NCOL + col] = acc[j] + bcol;
}

// ---------------------------------------------------------------------------
// Phase 2 helpers
// ---------------------------------------------------------------------------
__device__ __forceinline__ float fsig(float x)  { return 1.0f / (1.0f + __expf(-x)); }
__device__ __forceinline__ float ftanh(float x) { return 2.0f / (1.0f + __expf(-2.0f * x)) - 1.0f; }

// One 16x16 gate tile: acc = h_tile(16x256, LDS bf16) @ Wh_tile(256x16, global bf16)
__device__ __forceinline__ v8f gate_tile(const bf16_t* __restrict__ aBase,
                                         const bf16_t* __restrict__ bBase) {
    v8f acc = {};
#pragma unroll
    for (int kk = 0; kk < 8; ++kk) {
        const int k0 = kk * 32;
        v8bf aL0 = *(const v8bf*)(aBase + k0);         // ds_load_b128
        v8bf aL1 = *(const v8bf*)(aBase + k0 + 16);
        v16bf a = __builtin_shufflevector(aL0, aL1, 0,1,2,3,4,5,6,7,8,9,10,11,12,13,14,15);
        v8bf b0 = *(const v8bf*)(bBase + k0);
        v8bf b1 = *(const v8bf*)(bBase + k0 + 16);
        v16bf b = __builtin_shufflevector(b0, b1, 0,1,2,3,4,5,6,7,8,9,10,11,12,13,14,15);
        acc = __builtin_amdgcn_wmma_f32_16x16x32_bf16(false, a, false, b, (short)0, acc, false, false);
    }
    return acc;
}

// ---------------------------------------------------------------------------
// Phase 2: sequential scan over CHUNK_T steps; single workgroup (16 waves).
// h double-buffered in LDS (bf16); c held entirely in VGPRs.
// ---------------------------------------------------------------------------
__global__ __launch_bounds__(512) void lstm_recurrent_kernel(
        const float* __restrict__ Z,        // [chunkT*BATCH][NCOL]  (x-proj + bias)
        const bf16_t* __restrict__ Whp,     // [NCOL][DH]
        bf16_t* __restrict__ hstate,        // [BATCH*DH]
        float* __restrict__ cstate,         // [BATCH*DH]
        float* __restrict__ outs,           // [chunkT][BATCH][DH]
        int chunkT) {
    extern __shared__ bf16_t hbuf[];        // 2 * BATCH*DH bf16
    bf16_t* hcur  = hbuf;
    bf16_t* hnext = hbuf + BATCH * DH;

    const int tid  = threadIdx.x;
    const int lane = tid & 31;
    const int wave = tid >> 5;              // 0..15
    const int l16  = lane & 15;
    const int hi   = lane >> 4;

    for (int i = tid; i < BATCH * DH; i += 512) hcur[i] = hstate[i];

    // wave w owns h-columns w*16..w*16+15 (ntile = w) for all 4 batch m-tiles
    float c[4][8];
#pragma unroll
    for (int p = 0; p < 4; ++p) {
        const int pos  = wave * 4 + p;
        const int mt   = pos & 3;
        const int nt   = pos >> 2;
        const int colh = nt * 16 + l16;
#pragma unroll
        for (int j = 0; j < 8; ++j)
            c[p][j] = cstate[(mt * 16 + hi * 8 + j) * DH + colh];
    }
    __syncthreads();

#pragma unroll 1
    for (int t = 0; t < chunkT; ++t) {
        const float* Zt   = Z    + (size_t)t * BATCH * NCOL;
        float*       outT = outs + (size_t)t * BATCH * DH;
#pragma unroll
        for (int p = 0; p < 4; ++p) {
            const int pos  = wave * 4 + p;
            const int mt   = pos & 3;
            const int nt   = pos >> 2;
            const int colh = nt * 16 + l16;
            const int rowB = mt * 16 + hi * 8;
            const bf16_t* aBase = hcur + (mt * 16 + l16) * DH + hi * 8;

            v8f af = gate_tile(aBase, Whp + (0 * DH + colh) * DH + hi * 8);
            v8f ai = gate_tile(aBase, Whp + (1 * DH + colh) * DH + hi * 8);
            v8f ag = gate_tile(aBase, Whp + (2 * DH + colh) * DH + hi * 8);
            v8f ao = gate_tile(aBase, Whp + (3 * DH + colh) * DH + hi * 8);

#pragma unroll
            for (int j = 0; j < 8; ++j) {
                const int row = rowB + j;
                const float pf = af[j] + Zt[row * NCOL + (0 * DH + colh)];
                const float pi = ai[j] + Zt[row * NCOL + (1 * DH + colh)];
                const float pg = ag[j] + Zt[row * NCOL + (2 * DH + colh)];
                const float po = ao[j] + Zt[row * NCOL + (3 * DH + colh)];
                const float fv = fsig(pf);
                const float iv = fsig(pi);
                const float gv = ftanh(pg);
                const float ov = fsig(po);
                const float cn = fv * c[p][j] + iv * gv;
                c[p][j] = cn;
                const float hv = ov * ftanh(cn);
                __builtin_nontemporal_store(hv, &outT[row * DH + colh]);  // stream-out
                hnext[row * DH + colh] = (bf16_t)hv;                      // next-step A
            }
        }
        bf16_t* tmp = hcur; hcur = hnext; hnext = tmp;
        __syncthreads();                                // hnext complete before next reads
    }

    for (int i = tid; i < BATCH * DH; i += 512) hstate[i] = hcur[i];
#pragma unroll
    for (int p = 0; p < 4; ++p) {
        const int pos  = wave * 4 + p;
        const int mt   = pos & 3;
        const int nt   = pos >> 2;
        const int colh = nt * 16 + l16;
#pragma unroll
        for (int j = 0; j < 8; ++j)
            cstate[(mt * 16 + hi * 8 + j) * DH + colh] = c[p][j];
    }
}

// hT = outs[T-1], cT = cstate  -> appended to d_out
__global__ void finalize_kernel(const float* __restrict__ cstate, float* __restrict__ out) {
    int i = blockIdx.x * blockDim.x + threadIdx.x;
    if (i < BATCH * DH) {
        float* tail = out + (size_t)T_STEPS * BATCH * DH;
        tail[i]              = out[(size_t)(T_STEPS - 1) * BATCH * DH + i];
        tail[BATCH * DH + i] = cstate[i];
    }
}

// ---------------------------------------------------------------------------
extern "C" void kernel_launch(void* const* d_in, const int* in_sizes, int n_in,
                              void* d_out, int out_size, void* d_ws, size_t ws_size,
                              hipStream_t stream) {
    const float* X   = (const float*)d_in[0];
    const float* Wf  = (const float*)d_in[1];
    const float* bf_ = (const float*)d_in[2];
    const float* Wi  = (const float*)d_in[3];
    const float* bi_ = (const float*)d_in[4];
    const float* Wg  = (const float*)d_in[5];
    const float* bg_ = (const float*)d_in[6];
    const float* Wo  = (const float*)d_in[7];
    const float* bo_ = (const float*)d_in[8];
    float* out = (float*)d_out;

    char* ws = (char*)d_ws;
    size_t off = 0;
    bf16_t* Wxp    = (bf16_t*)(ws + off); off += (size_t)NCOL * DH * sizeof(bf16_t);
    bf16_t* Whp    = (bf16_t*)(ws + off); off += (size_t)NCOL * DH * sizeof(bf16_t);
    float*  bias   = (float*) (ws + off); off += (size_t)NCOL * sizeof(float);
    bf16_t* hstate = (bf16_t*)(ws + off); off += (size_t)BATCH * DH * sizeof(bf16_t);
    off = (off + 255) & ~(size_t)255;
    float*  cstate = (float*) (ws + off); off += (size_t)BATCH * DH * sizeof(float);
    off = (off + 255) & ~(size_t)255;
    bf16_t* Xb     = (bf16_t*)(ws + off); off += (size_t)CHUNK_ROWS * DH * sizeof(bf16_t); // 4 MB
    off = (off + 255) & ~(size_t)255;
    float*  Z      = (float*) (ws + off); // CHUNK_ROWS * NCOL f32 = 32 MB

    pack_weights_kernel<<<(NCOL * DH) / 256, 256, 0, stream>>>(Wf, Wi, Wg, Wo, bf_, bi_, bg_, bo_,
                                                               Wxp, Whp, bias);
    zero_state_kernel<<<(BATCH * DH) / 256, 256, 0, stream>>>(hstate, cstate);

    for (int ch = 0; ch < NCHUNK; ++ch) {
        const float* Xc = X + (size_t)ch * CHUNK_ROWS * DH;   // D_IN == 256
        xconv_kernel<<<(CHUNK_ROWS * DH / 8) / 256, 256, 0, stream>>>(Xc, Xb);
        xproj_kernel<<<dim3(CHUNK_ROWS / 16, 8), 256, 0, stream>>>(Xb, Wxp, bias, Z);
        lstm_recurrent_kernel<<<1, 512, 2 * BATCH * DH * sizeof(bf16_t), stream>>>(
            Z, Whp, hstate, cstate, out + (size_t)ch * CHUNK_ROWS * DH, CHUNK_T);
    }
    finalize_kernel<<<(BATCH * DH + 255) / 256, 256, 0, stream>>>(cstate, out);
}
